// ScaledDotProductAttention_80831284510747
// MI455X (gfx1250) — matrix operand order repaired and checked by reading, not compiled, not run on hardware
//
#include <hip/hip_runtime.h>

typedef __attribute__((ext_vector_type(16))) _Float16 v16h;
typedef __attribute__((ext_vector_type(8)))  _Float16 v8h;
typedef __attribute__((ext_vector_type(8)))  float    v8f;

#define B_   16
#define LQ_  2048
#define LK_  2048
#define DK_  128
#define DV_  128
#define BQ   64
#define BK   64
#define PAD  8

__global__ __launch_bounds__(128)
void fa_fwd_causal(const float* __restrict__ Qg,
                   const float* __restrict__ Kg,
                   const float* __restrict__ Vg,
                   float* __restrict__ Og) {
  // LDS staging (f16), padded to avoid bank conflicts on row-strided reads
  __shared__ __align__(16) _Float16 sK [BK][DK_ + PAD];   // K tile, row-major [key][d]
  __shared__ __align__(16) _Float16 sVT[DV_][BK + PAD];   // V tile transposed [dv][key]
  __shared__ __align__(16) _Float16 sP [4][16][BK + PAD]; // per-wave P scratch [wave][m][key]

  const int tid  = threadIdx.x;
  const int wv   = tid >> 5;
  const int lane = tid & 31;
  const int ln   = lane & 15;   // N index / row-in-halfwave
  const int hi   = lane >> 4;   // half-wave select

  const int qblk = blockIdx.x;
  const int b    = blockIdx.y;
  const int q0   = qblk * BQ;

  const float inv_temp = 0.0883883476483184f; // 1/sqrt(128)

  // ---- Pre-pack Q rows (A-matrix layout), 4 k-steps of 32 covering DK=128
  v16h qf[4];
  {
    const int qrow = q0 + wv * 16 + ln;
    const float* qp = Qg + ((size_t)b * LQ_ + qrow) * DK_;
#pragma unroll
    for (int s = 0; s < 4; ++s) {
      const int d1 = 32 * s + 8 * hi;        // halves 0..7  -> d = d1+e
      const int d2 = 32 * s + 16 + 8 * hi;   // halves 8..15 -> d = d2+e
      const float4 f0 = *(const float4*)(qp + d1);
      const float4 f1 = *(const float4*)(qp + d1 + 4);
      const float4 f2 = *(const float4*)(qp + d2);
      const float4 f3 = *(const float4*)(qp + d2 + 4);
      qf[s][0]=(_Float16)f0.x;  qf[s][1]=(_Float16)f0.y;  qf[s][2]=(_Float16)f0.z;  qf[s][3]=(_Float16)f0.w;
      qf[s][4]=(_Float16)f1.x;  qf[s][5]=(_Float16)f1.y;  qf[s][6]=(_Float16)f1.z;  qf[s][7]=(_Float16)f1.w;
      qf[s][8]=(_Float16)f2.x;  qf[s][9]=(_Float16)f2.y;  qf[s][10]=(_Float16)f2.z; qf[s][11]=(_Float16)f2.w;
      qf[s][12]=(_Float16)f3.x; qf[s][13]=(_Float16)f3.y; qf[s][14]=(_Float16)f3.z; qf[s][15]=(_Float16)f3.w;
    }
  }

  // ---- Accumulators: O (16x128 f32, D-layout) + online-softmax row state
  v8f o[8];
  v8f vzero = {};
#pragma unroll
  for (int nt = 0; nt < 8; ++nt) o[nt] = vzero;
  float rm[8], rl[8];
#pragma unroll
  for (int r = 0; r < 8; ++r) { rm[r] = -1e30f; rl[r] = 0.0f; }

  const int ntiles = qblk + 1;   // causal: keys 0 .. q0+63 only
  for (int kt = 0; kt < ntiles; ++kt) {
    const int kb = kt * BK;

    __syncthreads();  // previous tile fully consumed before overwrite

    // ---- cooperative load: K tile -> f16 LDS (row-major)
    for (int i = tid; i < BK * (DK_ / 4); i += 128) {
      const int row = i >> 5;          // 32 float4 per row
      const int c4  = i & 31;
      const float4 f = *(const float4*)(Kg + ((size_t)b * LK_ + kb + row) * DK_ + c4 * 4);
      _Float16* d = &sK[row][c4 * 4];
      d[0] = (_Float16)f.x; d[1] = (_Float16)f.y; d[2] = (_Float16)f.z; d[3] = (_Float16)f.w;
    }
    // ---- cooperative load: V tile -> f16 LDS, transposed [dv][key]
    for (int i = tid; i < BK * (DV_ / 4); i += 128) {
      const int row = i >> 5;
      const int c4  = i & 31;
      const float4 f = *(const float4*)(Vg + ((size_t)b * LK_ + kb + row) * DV_ + c4 * 4);
      sVT[c4 * 4 + 0][row] = (_Float16)f.x;
      sVT[c4 * 4 + 1][row] = (_Float16)f.y;
      sVT[c4 * 4 + 2][row] = (_Float16)f.z;
      sVT[c4 * 4 + 3][row] = (_Float16)f.w;
    }
    __syncthreads();

    // ---- S = Q * K^T  (4 sub-tiles of 16 keys, K-dim 128 = 4 wmma steps)
    v8f S[4];
#pragma unroll
    for (int t = 0; t < 4; ++t) {
      S[t] = vzero;
#pragma unroll
      for (int s = 0; s < 4; ++s) {
        // B-matrix layout: lane%16 = key n, halves h<8 -> d = 32s + 8*hi + h, h>=8 -> d = 32s + 16 + 8*hi + (h-8)
        const _Float16* bp = &sK[t * 16 + ln][32 * s + 8 * hi];
        const v8h blo = *(const v8h*)bp;
        const v8h bhi = *(const v8h*)(bp + 16);
        v16h bf;
#pragma unroll
        for (int e = 0; e < 8; ++e) { bf[e] = blo[e]; bf[e + 8] = bhi[e]; }
        S[t] = __builtin_amdgcn_wmma_f32_16x16x32_f16(false, qf[s], false, bf,
                                                      (short)0, S[t], false, false);
      }
    }

    // ---- scale + causal mask + row max (this tile)
    float tmax[8];
#pragma unroll
    for (int r = 0; r < 8; ++r) {
      const int qi = q0 + wv * 16 + r + 8 * hi;
      float mx = -1e30f;
#pragma unroll
      for (int t = 0; t < 4; ++t) {
        const int kj = kb + t * 16 + ln;
        float sv = S[t][r] * inv_temp;
        sv = (kj > qi) ? -1e30f : sv;
        S[t][r] = sv;
        mx = fmaxf(mx, sv);
      }
#pragma unroll
      for (int off = 1; off < 16; off <<= 1)
        mx = fmaxf(mx, __shfl_xor(mx, off, 16));
      tmax[r] = mx;
    }

    // ---- online softmax update; P -> wave-private LDS (D-layout -> memory)
#pragma unroll
    for (int r = 0; r < 8; ++r) {
      const float mnew  = fmaxf(rm[r], tmax[r]);
      const float alpha = __expf(rm[r] - mnew);
      float psum = 0.0f;
#pragma unroll
      for (int t = 0; t < 4; ++t) {
        const float p = __expf(S[t][r] - mnew);
        psum += p;
        sP[wv][r + 8 * hi][t * 16 + ln] = (_Float16)p;
      }
#pragma unroll
      for (int off = 1; off < 16; off <<= 1)
        psum += __shfl_xor(psum, off, 16);
      rl[r] = rl[r] * alpha + psum;
      rm[r] = mnew;
#pragma unroll
      for (int nt = 0; nt < 8; ++nt) o[nt][r] *= alpha;
    }

    // ---- O += P * V   (K-dim 64 = 2 wmma steps; 8 output N-tiles of 16)
    // LDS is in-order within a wave: sP store->load needs no barrier.
#pragma unroll
    for (int s = 0; s < 2; ++s) {
      const _Float16* pp = &sP[wv][ln][0];
      const int k1 = 32 * s + 8 * hi;
      const int k2 = 32 * s + 16 + 8 * hi;
      const v8h a0 = *(const v8h*)(pp + k1);
      const v8h a1 = *(const v8h*)(pp + k2);
      v16h af;
#pragma unroll
      for (int e = 0; e < 8; ++e) { af[e] = a0[e]; af[e + 8] = a1[e]; }
#pragma unroll
      for (int nt = 0; nt < 8; ++nt) {
        const _Float16* vp = &sVT[nt * 16 + ln][32 * s + 8 * hi];
        const v8h b0 = *(const v8h*)vp;
        const v8h b1 = *(const v8h*)(vp + 16);
        v16h bf;
#pragma unroll
        for (int e = 0; e < 8; ++e) { bf[e] = b0[e]; bf[e + 8] = b1[e]; }
        o[nt] = __builtin_amdgcn_wmma_f32_16x16x32_f16(false, af, false, bf,
                                                       (short)0, o[nt], false, false);
      }
    }
  }

  // ---- epilogue: normalize and store fp32
  const int qbase = q0 + wv * 16;
#pragma unroll
  for (int r = 0; r < 8; ++r) {
    const float inv = 1.0f / rl[r];
    const int m = r + 8 * hi;
    float* op = Og + ((size_t)b * LQ_ + qbase + m) * DV_ + ln;
#pragma unroll
    for (int nt = 0; nt < 8; ++nt)
      op[nt * 16] = o[nt][r] * inv;
  }
}

extern "C" void kernel_launch(void* const* d_in, const int* in_sizes, int n_in,
                              void* d_out, int out_size, void* d_ws, size_t ws_size,
                              hipStream_t stream) {
  const float* Q = (const float*)d_in[0];
  const float* K = (const float*)d_in[1];
  const float* V = (const float*)d_in[2];
  // d_in[3] is the causal mask; recomputed analytically in-kernel.
  float* O = (float*)d_out;
  (void)in_sizes; (void)n_in; (void)out_size; (void)d_ws; (void)ws_size;

  dim3 grid(LQ_ / BQ, B_, 1);
  dim3 block(128, 1, 1);
  fa_fwd_causal<<<grid, block, 0, stream>>>(Q, K, V, O);
}
